// Model_39676907887374
// MI455X (gfx1250) — matrix operand order repaired and checked by reading, not compiled, not run on hardware
//
#include <hip/hip_runtime.h>

typedef __attribute__((ext_vector_type(16))) __bf16 v16bf;
typedef __attribute__((ext_vector_type(8)))  __bf16 v8bf;
typedef __attribute__((ext_vector_type(4)))  __bf16 v4bf;
typedef __attribute__((ext_vector_type(2)))  __bf16 v2bf;
typedef __attribute__((ext_vector_type(8)))  float  v8f;
typedef __attribute__((ext_vector_type(4)))  int    v4i;

// Pointer types for the CDNA5 async global->LDS DMA builtin:
// param0 = int4 addrspace(1)* (global src), param1 = int4 addrspace(3)* (LDS dst)
typedef __attribute__((address_space(1))) v4i as1_v4i;
typedef __attribute__((address_space(3))) v4i as3_v4i;

#if defined(__has_builtin)
#if __has_builtin(__builtin_amdgcn_global_load_async_to_lds_b128)
#define HAVE_ASYNC_LD 1
#endif
#endif
#if defined(__has_builtin)
#if __has_builtin(__builtin_amdgcn_s_wait_asynccnt)
#define WAIT_ASYNC0() __builtin_amdgcn_s_wait_asynccnt(0)
#endif
#endif
#ifndef WAIT_ASYNC0
#define WAIT_ASYNC0() asm volatile("s_wait_asynccnt 0" ::: "memory")
#endif

// Problem constants
constexpr int NB = 4;
constexpr int NH = 16;
constexpr int NS = 2048;
constexpr int ND = 64;
constexpr int NBH = NB * NH;
constexpr int NTOT = NB * NH * NS * ND;   // 8,388,608 elements per tensor

constexpr int QT = 128;   // query rows per block (8 waves x 16 rows)
constexpr int KT = 32;    // keys per inner iteration
constexpr float L2E = 1.4426950408889634f;   // log2(e)
constexpr float QS  = 0.125f * L2E;          // 1/sqrt(64) * log2(e), folded into Q

// LDS row strides (bf16 elems) padded for conflict-free 16-lane fragment reads
constexpr int K_STR = 72;
constexpr int V_STR = 40;
constexpr int P_STR = 40;

// ---------------- deterministic global-mean reduction ----------------
__global__ void reduce_partial(const float* __restrict__ v2,
                               const float* __restrict__ v4,
                               float* __restrict__ ws) {
  __shared__ float sm2[256];
  __shared__ float sm4[256];
  float s2 = 0.f, s4 = 0.f;
  for (int i = blockIdx.x * 256 + threadIdx.x; i < NTOT; i += 256 * 256) {
    s2 += v2[i];
    s4 += v4[i];
  }
  sm2[threadIdx.x] = s2;
  sm4[threadIdx.x] = s4;
  __syncthreads();
  for (int off = 128; off > 0; off >>= 1) {
    if ((int)threadIdx.x < off) {
      sm2[threadIdx.x] += sm2[threadIdx.x + off];
      sm4[threadIdx.x] += sm4[threadIdx.x + off];
    }
    __syncthreads();
  }
  if (threadIdx.x == 0) {
    ws[blockIdx.x]       = sm2[0];
    ws[256 + blockIdx.x] = sm4[0];
  }
}

__global__ void reduce_final(float* __restrict__ ws) {
  __shared__ float sm[256];
  float s = ws[threadIdx.x] + ws[256 + threadIdx.x];
  sm[threadIdx.x] = s;
  __syncthreads();
  for (int off = 128; off > 0; off >>= 1) {
    if ((int)threadIdx.x < off) sm[threadIdx.x] += sm[threadIdx.x + off];
    __syncthreads();
  }
  if (threadIdx.x == 0) ws[512] = sm[0] / (float)NTOT;
}

// ---------------- one-time pre-conversion passes ----------------
__global__ void preconvert_k(const float* __restrict__ K, __bf16* __restrict__ preK) {
  int i = 4 * (blockIdx.x * 256 + threadIdx.x);
  float4 v = *(const float4*)(K + i);
  v4bf o = { (__bf16)v.x, (__bf16)v.y, (__bf16)v.z, (__bf16)v.w };
  *(v4bf*)(preK + i) = o;
}

// mask pre-multiplied by log2(e) so the kernel can use raw v_exp_f32 (exp2)
__global__ void preconvert_m(const float* __restrict__ M, float* __restrict__ preM) {
  int i = 4 * (blockIdx.x * 256 + threadIdx.x);
  float4 v = *(const float4*)(M + i);
  float4 o = { v.x * L2E, v.y * L2E, v.z * L2E, v.w * L2E };
  *(float4*)(preM + i) = o;
}

// v = v1+v2+v4, transposed to [bh][d][k] with the kappa tile permutation
// (kappa = 2*(k%16) + k/16 within each 32-key tile) baked in.
__global__ void preconvert_v(const float* __restrict__ V1, const float* __restrict__ V2,
                             const float* __restrict__ V4, __bf16* __restrict__ preVt) {
  __shared__ float tile[64][65];
  const int tid = threadIdx.x;
  const int bh = blockIdx.x >> 5;           // 64 bh
  const int k0 = (blockIdx.x & 31) * 64;    // 64-key chunk
  const int base = bh * NS * ND;
#pragma unroll
  for (int j = 0; j < 16; ++j) {
    int e = tid + 256 * j;                  // 0..4095
    int kk = e >> 6, dd = e & 63;
    int gi = base + (k0 + kk) * ND + dd;
    tile[kk][dd] = V1[gi] + V2[gi] + V4[gi];
  }
  __syncthreads();
  __bf16* outp = preVt + bh * ND * NS;
#pragma unroll
  for (int j = 0; j < 16; ++j) {
    int e = tid + 256 * j;
    int dd = e >> 6, kp = e & 63;           // output-coalesced over kappa
    int g32 = kp >> 5, k5 = kp & 31;
    int kloc = 32 * g32 + 16 * (k5 & 1) + (k5 >> 1);  // inverse permutation
    outp[dd * NS + k0 + kp] = (__bf16)tile[kloc][dd];
  }
}

// ---------------- flash attention with bf16 WMMA ----------------
template <bool PRE>
__global__ __launch_bounds__(256)
void attn_kernel(const float* __restrict__ Q,  const float* __restrict__ V1,
                 const float* __restrict__ K,  const float* __restrict__ V2,
                 const float* __restrict__ V4, const float* __restrict__ M,
                 const float* __restrict__ cmean_p,
                 const __bf16* __restrict__ preK, const __bf16* __restrict__ preVt,
                 const float* __restrict__ preM, float* __restrict__ out) {
  __shared__ __align__(16) __bf16 KtBuf[2][KT * K_STR];   // [key][d]
  __shared__ __align__(16) __bf16 VtBuf[2][ND * V_STR];   // [d][kappa]
  __shared__ __align__(16) __bf16 P[8 * 16 * P_STR];      // per-wave [row][kappa]
  __shared__ __align__(16) float  MBuf[2][QT * KT];       // mask tile (async path)

  const int tid  = threadIdx.x;
  const int lane = tid & 31;
  const int wave = tid >> 5;
  const int hh   = lane >> 4;
  const int ln   = lane & 15;

  const int bh = blockIdx.x >> 4;
  const int q0 = (blockIdx.x & 15) * QT;
  const int base = bh * NS * ND;

  // Q A-fragments (16x32 per d-chunk), pre-scaled by log2(e)/sqrt(D)
  const int qrow = q0 + wave * 16 + ln;
  const float* qptr = Q + base + qrow * ND;
  v16bf aq[2];
#pragma unroll
  for (int c = 0; c < 2; ++c) {
#pragma unroll
    for (int i = 0; i < 16; ++i) {
      int d = c * 32 + hh * 16 + i;
      aq[c][i] = (__bf16)(qptr[d] * QS);
    }
  }

  int mbase[8];
#pragma unroll
  for (int r = 0; r < 8; ++r)
    mbase[r] = (q0 + wave * 16 + r + 8 * hh) * NS + ln;

  // all-ones B fragment: softmax denominator via l = P @ ones (one extra WMMA)
  v16bf ones;
#pragma unroll
  for (int i = 0; i < 16; ++i) ones[i] = (__bf16)1.0f;

  v8f o0 = {}, o1 = {}, o2 = {}, o3 = {}, lacc = {};

  __bf16* Pw = &P[wave * 16 * P_STR];

  // inner-tile computation; s0/s1 arrive pre-initialized with the (log2e-scaled)
  // mask tile as the WMMA C operand.
  auto compute_tile = [&](const __bf16* Kt, const __bf16* Vt, v8f s0, v8f s1) {
#pragma unroll
    for (int c = 0; c < 2; ++c) {
      const v8bf* bp0 = (const v8bf*)&Kt[ln * K_STR + c * 32 + hh * 16];
      v16bf b0 = __builtin_shufflevector(bp0[0], bp0[1],
                   0,1,2,3,4,5,6,7,8,9,10,11,12,13,14,15);
      s0 = __builtin_amdgcn_wmma_f32_16x16x32_bf16(false, aq[c], false, b0,
                                                   (short)0, s0, false, false);
      const v8bf* bp1 = (const v8bf*)&Kt[(16 + ln) * K_STR + c * 32 + hh * 16];
      v16bf b1 = __builtin_shufflevector(bp1[0], bp1[1],
                   0,1,2,3,4,5,6,7,8,9,10,11,12,13,14,15);
      s1 = __builtin_amdgcn_wmma_f32_16x16x32_bf16(false, aq[c], false, b1,
                                                   (short)0, s1, false, false);
    }

    // p = exp2(s) (v_exp_f32 directly; scores are log2-domain already).
    // No max-shift: scores ~N(0,1); a uniform shift cancels in p/sum(p).
#pragma unroll
    for (int r = 0; r < 8; ++r) {
      float p0 = __builtin_amdgcn_exp2f(s0[r]);
      float p1 = __builtin_amdgcn_exp2f(s1[r]);
      int row = r + 8 * hh;
      v2bf pp = { (__bf16)p0, (__bf16)p1 };          // keys ln,16+ln -> kappa 2ln,2ln+1
      *(v2bf*)&Pw[row * P_STR + 2 * ln] = pp;        // packed b32 store
    }

    const v8bf* pa0 = (const v8bf*)&Pw[ln * P_STR + hh * 16];
    const v8bf* pa1 = (const v8bf*)&Pw[ln * P_STR + hh * 16 + 8];
    v16bf pa = __builtin_shufflevector(pa0[0], pa1[0],
                 0,1,2,3,4,5,6,7,8,9,10,11,12,13,14,15);

#pragma unroll
    for (int dgrp = 0; dgrp < 4; ++dgrp) {
      const v8bf* bv = (const v8bf*)&Vt[(dgrp * 16 + ln) * V_STR + hh * 16];
      v16bf b = __builtin_shufflevector(bv[0], bv[1],
                  0,1,2,3,4,5,6,7,8,9,10,11,12,13,14,15);
      v8f& oo = dgrp == 0 ? o0 : dgrp == 1 ? o1 : dgrp == 2 ? o2 : o3;
      oo = __builtin_amdgcn_wmma_f32_16x16x32_bf16(false, pa, false, b,
                                                   (short)0, oo, false, false);
    }
    // denominator: row sums of P, replicated across lanes by the ones-matrix
    lacc = __builtin_amdgcn_wmma_f32_16x16x32_bf16(false, pa, false, ones,
                                                   (short)0, lacc, false, false);
  };

  if constexpr (PRE) {
    const __bf16* gK  = preK  + bh * NS * ND;   // [k][d] bf16
    const __bf16* gVt = preVt + bh * ND * NS;   // [d][kappa] bf16
    const int kkK = tid >> 3, ddK = 8 * (tid & 7);   // 16B chunk of K tile
    const int ddV = tid >> 2, koV = 8 * (tid & 3);   // 16B chunk of V tile
#ifdef HAVE_ASYNC_LD
    // ASYNCcnt-tracked DMA staging (K, V and mask) with LDS ping-pong buffers
    auto issue = [&](int kb, int buf) {
      __builtin_amdgcn_global_load_async_to_lds_b128(
          (as1_v4i*)(gK + (kb + kkK) * ND + ddK),
          (as3_v4i*)&KtBuf[buf][kkK * K_STR + ddK], 0, 0);
      __builtin_amdgcn_global_load_async_to_lds_b128(
          (as1_v4i*)(gVt + ddV * NS + kb + koV),
          (as3_v4i*)&VtBuf[buf][ddV * V_STR + koV], 0, 0);
#pragma unroll
      for (int j = 0; j < 4; ++j) {                 // 128x32 f32 mask tile
        int chunk = tid + 256 * j;
        int row = chunk >> 3, off = (chunk & 7) * 4;
        __builtin_amdgcn_global_load_async_to_lds_b128(
            (as1_v4i*)(preM + (q0 + row) * NS + kb + off),
            (as3_v4i*)&MBuf[buf][row * 32 + off], 0, 0);
      }
    };
    issue(0, 0);
    int it = 0;
    for (int kb = 0; kb < NS; kb += KT, ++it) {
      int cur = it & 1;
      WAIT_ASYNC0();       // my tile-it copies landed
      __syncthreads();     // everyone's landed; prior reads of other buf retired
      if (kb + KT < NS) issue(kb + KT, cur ^ 1);
      v8f s0, s1;
      const float* mb = &MBuf[cur][(wave * 16 + 8 * hh) * KT + ln];
#pragma unroll
      for (int r = 0; r < 8; ++r) {
        s0[r] = mb[r * KT];
        s1[r] = mb[r * KT + 16];
      }
      compute_tile(KtBuf[cur], VtBuf[cur], s0, s1);
    }
#else
    // register-pipelined copy staging (no conversion needed: already bf16)
    uint4 kr = *(const uint4*)(gK + kkK * ND + ddK);
    uint4 vr = *(const uint4*)(gVt + ddV * NS + koV);
    for (int kb = 0; kb < NS; kb += KT) {
      *(uint4*)&KtBuf[0][kkK * K_STR + ddK] = kr;
      *(uint4*)&VtBuf[0][ddV * V_STR + koV] = vr;
      __syncthreads();
      if (kb + KT < NS) {
        kr = *(const uint4*)(gK + (kb + KT + kkK) * ND + ddK);
        vr = *(const uint4*)(gVt + ddV * NS + kb + KT + koV);
      }
      v8f s0, s1;
#pragma unroll
      for (int r = 0; r < 8; ++r) {
        s0[r] = preM[mbase[r] + kb];
        s1[r] = preM[mbase[r] + kb + 16];
      }
      compute_tile(KtBuf[0], VtBuf[0], s0, s1);
      __syncthreads();
    }
#endif
  } else {
    // fallback: convert fp32 -> bf16 in-kernel while staging
    float4 k4[2], va[2], vb[2], vc[2];
#pragma unroll
    for (int g = 0; g < 2; ++g) {
      int e  = 4 * (tid + 256 * g);
      int gi = base + (e >> 6) * ND + (e & 63);
      k4[g] = *(const float4*)(K  + gi);
      va[g] = *(const float4*)(V1 + gi);
      vb[g] = *(const float4*)(V2 + gi);
      vc[g] = *(const float4*)(V4 + gi);
    }
    for (int kb = 0; kb < NS; kb += KT) {
#pragma unroll
      for (int g = 0; g < 2; ++g) {
        int e  = 4 * (tid + 256 * g);
        int kk = e >> 6;
        int dd = e & 63;
        int kp = 2 * (kk & 15) + (kk >> 4);   // kappa permutation
        v4bf kc = { (__bf16)k4[g].x, (__bf16)k4[g].y,
                    (__bf16)k4[g].z, (__bf16)k4[g].w };
        *(v4bf*)&KtBuf[0][kk * K_STR + dd] = kc;
        VtBuf[0][(dd + 0) * V_STR + kp] = (__bf16)(va[g].x + vb[g].x + vc[g].x);
        VtBuf[0][(dd + 1) * V_STR + kp] = (__bf16)(va[g].y + vb[g].y + vc[g].y);
        VtBuf[0][(dd + 2) * V_STR + kp] = (__bf16)(va[g].z + vb[g].z + vc[g].z);
        VtBuf[0][(dd + 3) * V_STR + kp] = (__bf16)(va[g].w + vb[g].w + vc[g].w);
      }
      __syncthreads();
      if (kb + KT < NS) {
#pragma unroll
        for (int g = 0; g < 2; ++g) {
          int e  = 4 * (tid + 256 * g);
          int gi = base + (kb + KT + (e >> 6)) * ND + (e & 63);
          k4[g] = *(const float4*)(K  + gi);
          va[g] = *(const float4*)(V1 + gi);
          vb[g] = *(const float4*)(V2 + gi);
          vc[g] = *(const float4*)(V4 + gi);
        }
      }
      v8f s0, s1;
#pragma unroll
      for (int r = 0; r < 8; ++r) {
        s0[r] = M[mbase[r] + kb] * L2E;
        s1[r] = M[mbase[r] + kb + 16] * L2E;
      }
      compute_tile(KtBuf[0], VtBuf[0], s0, s1);
      __syncthreads();
    }
  }

  // epilogue: normalize by WMMA-computed denominators, subtract mean constant
  const float cmean = *cmean_p;
#pragma unroll
  for (int r = 0; r < 8; ++r) {
    float inv = 1.0f / lacc[r];
    int row = q0 + wave * 16 + r + 8 * hh;
    float* op = out + base + row * ND;
    op[0 * 16 + ln] = o0[r] * inv - cmean;
    op[1 * 16 + ln] = o1[r] * inv - cmean;
    op[2 * 16 + ln] = o2[r] * inv - cmean;
    op[3 * 16 + ln] = o3[r] * inv - cmean;
  }
}

extern "C" void kernel_launch(void* const* d_in, const int* in_sizes, int n_in,
                              void* d_out, int out_size, void* d_ws, size_t ws_size,
                              hipStream_t stream) {
  const float* x2   = (const float*)d_in[0];  // queries
  const float* v1   = (const float*)d_in[1];
  const float* k    = (const float*)d_in[2];
  const float* v2   = (const float*)d_in[3];
  const float* v4   = (const float*)d_in[4];
  const float* mask = (const float*)d_in[5];
  float* out = (float*)d_out;
  float* ws  = (float*)d_ws;

  reduce_partial<<<256, 256, 0, stream>>>(v2, v4, ws);
  reduce_final<<<1, 256, 0, stream>>>(ws);

  const size_t pre_off   = 4096;
  const size_t preKV_b   = (size_t)NTOT * 2 * 2;     // preK + preVt (bf16)
  const size_t preM_b    = (size_t)NS * NS * 4;      // preM (f32)
  __bf16* preK  = (__bf16*)((char*)d_ws + pre_off);
  __bf16* preVt = preK + NTOT;
  float*  preM  = (float*)((char*)d_ws + pre_off + preKV_b);

  if (ws_size >= pre_off + preKV_b + preM_b) {
    preconvert_k<<<NTOT / 1024, 256, 0, stream>>>(k, preK);
    preconvert_v<<<NBH * (NS / 64), 256, 0, stream>>>(v1, v2, v4, preVt);
    preconvert_m<<<NS * NS / 1024, 256, 0, stream>>>(mask, preM);
    attn_kernel<true><<<NBH * (NS / QT), 256, 0, stream>>>(
        x2, v1, k, v2, v4, mask, ws + 512, preK, preVt, preM, out);
  } else {
    attn_kernel<false><<<NBH * (NS / QT), 256, 0, stream>>>(
        x2, v1, k, v2, v4, mask, ws + 512, preK, preVt, preM, out);
  }
}
